// Model_5016521801872
// MI455X (gfx1250) — matrix operand order repaired and checked by reading, not compiled, not run on hardware
//
#include <hip/hip_runtime.h>

typedef __bf16 bf16_t;
typedef __attribute__((ext_vector_type(16))) __bf16 v16bf;
typedef __attribute__((ext_vector_type(8)))  float  v8f;

#define Bc    32
#define SEQc  336
#define ALLc  672
#define CINc  7
#define DMc   64
#define NFc   85
#define NFFTc 64
#define HWc   (NFFTc * NFc)              // 64*85 = 5440 (multiple of 32)
#define SBIG  ((size_t)Bc * DMc * HWc)   // 11,141,120 elements

// ---------------------------------------------------------------------------
// WMMA helpers
// ---------------------------------------------------------------------------
__device__ __forceinline__ v8f wmma_bf16(v16bf a, v16bf b, v8f c) {
  return __builtin_amdgcn_wmma_f32_16x16x32_bf16(false, a, false, b, (short)0, c,
                                                 false, false);
}

// A fragment: 16x32 tile of row-major [M x K]; branchless (clamp + select)
template <typename T>
__device__ __forceinline__ v16bf loadA(const T* A, long base, int m0, int k0,
                                       int M, int K, int lda, int lane) {
  v16bf f;
  int m = m0 + (lane & 15);
  m = m < M ? m : M - 1;                 // clamped rows feed unused D rows only
  const T* row = A + base + (long)m * lda;
  int kb = k0 + ((lane >> 4) << 4);
#pragma unroll
  for (int e = 0; e < 16; ++e) {
    int k = kb + e;
    int kc = k < K ? k : K - 1;
    float v = (float)row[kc];
    f[e] = (__bf16)(k < K ? v : 0.f);
  }
  return f;
}

// B fragment: 32x16 tile of [K x N]; trans reads Bt[n*ldb+k]; branchless
template <typename T>
__device__ __forceinline__ v16bf loadB(const T* Bp, long base, int k0, int n0,
                                       int K, int N, int ldb, int trans,
                                       int lane) {
  v16bf f;
  int n = n0 + (lane & 15);
  bool nok = (n < N);
  n = nok ? n : N - 1;
  int kb = k0 + ((lane >> 4) << 4);
  if (trans) {
    const T* row = Bp + base + (long)n * ldb;
#pragma unroll
    for (int e = 0; e < 16; ++e) {
      int k = kb + e;
      int kc = k < K ? k : K - 1;
      float v = (float)row[kc];
      f[e] = (__bf16)((nok && k < K) ? v : 0.f);
    }
  } else {
    const T* col = Bp + base + n;
#pragma unroll
    for (int e = 0; e < 16; ++e) {
      int k = kb + e;
      int kc = k < K ? k : K - 1;
      float v = (float)col[(long)kc * ldb];
      f[e] = (__bf16)((nok && k < K) ? v : 0.f);
    }
  }
  return f;
}

// Weight fragment for per-tap conv GEMM: k index = channel c, stride = taps
__device__ __forceinline__ v16bf loadWtap(const float* Wp, int o0, int c0,
                                          int taps, int tap, int Ktot,
                                          int lane) {
  v16bf f;
  int o = o0 + (lane & 15);              // O=64 exact, no clamp needed
  const float* row = Wp + (long)o * Ktot + tap;
  int cb = c0 + ((lane >> 4) << 4);
#pragma unroll
  for (int e = 0; e < 16; ++e) {
    int c = cb + e;
    f[e] = (__bf16)row[(long)c * taps];
  }
  return f;
}

// Input fragment for per-tap conv GEMM: lane-uniform offset/validity,
// elements strided by HW along the channel (=K) axis
__device__ __forceinline__ v16bf loadXtap(const bf16_t* X, long base, int c0,
                                          int off, bool valid, int lane) {
  v16bf f;
  const bf16_t* p = X + base + off;
  int cb = c0 + ((lane >> 4) << 4);
#pragma unroll
  for (int e = 0; e < 16; ++e) {
    float v = (float)p[(long)(cb + e) * HWc];
    f[e] = (__bf16)(valid ? v : 0.f);
  }
  return f;
}

// ---------------------------------------------------------------------------
// Generic dual-product batched GEMM:  C = A1*B1 + alpha2 * A2*B2  (A2 optional)
// ---------------------------------------------------------------------------
template <typename TA, typename TB>
__global__ __launch_bounds__(32)
void gemm_dual(const TA* A1, const TB* B1, const TA* A2, const TB* B2,
               float* Cf, bf16_t* Cb, const float* bias,
               int M, int N, int K, int lda, int ldb, int ldc,
               long sA, long sB, long sC,
               float alpha2, int transB, int biasMode, int epiSigmoid) {
  int lane = threadIdx.x;
  int ntn = (N + 15) >> 4;
  int tm = (blockIdx.x / ntn) << 4;
  int tn = (blockIdx.x % ntn) << 4;
  long bA = (long)blockIdx.y * sA;
  long bB = (long)blockIdx.y * sB;
  long bC = (long)blockIdx.y * sC;
  v8f acc1 = {};
  v8f acc2 = {};
  bool dual = (A2 != nullptr);
  for (int k0 = 0; k0 < K; k0 += 32) {
    v16bf a1 = loadA(A1, bA, tm, k0, M, K, lda, lane);
    v16bf b1 = loadB(B1, bB, k0, tn, K, N, ldb, transB, lane);
    acc1 = wmma_bf16(a1, b1, acc1);
    if (dual) {
      v16bf a2 = loadA(A2, bA, tm, k0, M, K, lda, lane);
      v16bf b2 = loadB(B2, bB, k0, tn, K, N, ldb, transB, lane);
      acc2 = wmma_bf16(a2, b2, acc2);
    }
  }
  int n = tn + (lane & 15);
  int mo = tm + ((lane >> 4) << 3);
#pragma unroll
  for (int v = 0; v < 8; ++v) {
    int m = mo + v;
    if (m < M && n < N) {
      float val = acc1[v] + (dual ? alpha2 * acc2[v] : 0.f);
      if (biasMode == 1) val += bias[m];
      else if (biasMode == 2) val += bias[n];
      if (epiSigmoid) val = 1.f / (1.f + __expf(-val));
      long idx = bC + (long)m * ldc + n;
      if (Cb) Cb[idx] = (__bf16)val;
      else    Cf[idx] = val;
    }
  }
}

// ---------------------------------------------------------------------------
// Complex conv, per-tap implicit GEMM. One wave = 16(out-ch) x 32(spatial).
//   out_r = Wr*Xr - Wi*Xi + (br-bi);  out_i = Wr*Xi + Wi*Xr + (br+bi)
// mode 0: write bf16 Or/Oi;  mode 1: SumR += relu(out_r), SumI += relu(out_i)
// O = C = 64, spatial = 64x85 = 5440: no edge tiles anywhere.
// ---------------------------------------------------------------------------
__global__ __launch_bounds__(32)
void cconv(const bf16_t* Xr, const bf16_t* Xi, const float* Wr, const float* Wi,
           const float* br, const float* bi, bf16_t* Or, bf16_t* Oi,
           float* SumR, float* SumI, int taps, int dil, int mode) {
  const int H = NFFTc, W = NFc, C = DMc;
  int lane = threadIdx.x;
  const int npair = HWc / 32;            // 170
  int to = (blockIdx.x / npair) << 4;
  int tn = (blockIdx.x % npair) << 5;
  int b = blockIdx.y;
  long xbase = (long)b * C * HWc;
  int Ktot = C * taps;

  int nn0 = tn + (lane & 15);
  int nn1 = nn0 + 16;
  int x0 = nn0 / W, w0 = nn0 - x0 * W;
  int x1 = nn1 / W, w1 = nn1 - x1 * W;

  v8f aRR0 = {}, aII0 = {}, aRI0 = {}, aIR0 = {};
  v8f aRR1 = {}, aII1 = {}, aRI1 = {}, aIR1 = {};

  for (int tap = 0; tap < taps; ++tap) {
    int ty = tap / 3;
    int dy = (taps == 9) ? (ty - 1) * dil : 0;
    int dx = (taps == 9) ? ((tap - ty * 3) - 1) * dil : 0;
    bool v0 = ((unsigned)(x0 + dy) < (unsigned)H) &&
              ((unsigned)(w0 + dx) < (unsigned)W);
    bool v1 = ((unsigned)(x1 + dy) < (unsigned)H) &&
              ((unsigned)(w1 + dx) < (unsigned)W);
    int off0 = v0 ? (x0 + dy) * W + (w0 + dx) : 0;
    int off1 = v1 ? (x1 + dy) * W + (w1 + dx) : 0;
    for (int c0 = 0; c0 < C; c0 += 32) {
      v16bf wr  = loadWtap(Wr, to, c0, taps, tap, Ktot, lane);
      v16bf wi  = loadWtap(Wi, to, c0, taps, tap, Ktot, lane);
      v16bf xr0 = loadXtap(Xr, xbase, c0, off0, v0, lane);
      v16bf xi0 = loadXtap(Xi, xbase, c0, off0, v0, lane);
      v16bf xr1 = loadXtap(Xr, xbase, c0, off1, v1, lane);
      v16bf xi1 = loadXtap(Xi, xbase, c0, off1, v1, lane);
      aRR0 = wmma_bf16(wr, xr0, aRR0);
      aII0 = wmma_bf16(wi, xi0, aII0);
      aRI0 = wmma_bf16(wr, xi0, aRI0);
      aIR0 = wmma_bf16(wi, xr0, aIR0);
      aRR1 = wmma_bf16(wr, xr1, aRR1);
      aII1 = wmma_bf16(wi, xi1, aII1);
      aRI1 = wmma_bf16(wr, xi1, aRI1);
      aIR1 = wmma_bf16(wi, xr1, aIR1);
    }
  }

  int mo = to + ((lane >> 4) << 3);
  long obase = (long)b * DMc * HWc;
#pragma unroll
  for (int v = 0; v < 8; ++v) {
    int o = mo + v;
    float bd = br[o] - bi[o];
    float bs = br[o] + bi[o];
    long idx0 = obase + (long)o * HWc + nn0;
    long idx1 = idx0 + 16;
    float rr0 = aRR0[v] - aII0[v] + bd;
    float ii0 = aRI0[v] + aIR0[v] + bs;
    float rr1 = aRR1[v] - aII1[v] + bd;
    float ii1 = aRI1[v] + aIR1[v] + bs;
    if (mode == 0) {
      Or[idx0] = (__bf16)rr0;  Oi[idx0] = (__bf16)ii0;
      Or[idx1] = (__bf16)rr1;  Oi[idx1] = (__bf16)ii1;
    } else {
      SumR[idx0] += fmaxf(rr0, 0.f);  SumI[idx0] += fmaxf(ii0, 0.f);
      SumR[idx1] += fmaxf(rr1, 0.f);  SumI[idx1] += fmaxf(ii1, 0.f);
    }
  }
}

// ---------------------------------------------------------------------------
// Small VALU kernels
// ---------------------------------------------------------------------------
__global__ void meanstd_k(const float* x, float* ms) {
  int bc = blockIdx.x;                 // B*CIN = 224
  int b = bc / CINc, c = bc % CINc;
  __shared__ float s1[64], s2[64];
  float a = 0.f, a2 = 0.f;
  for (int t = threadIdx.x; t < SEQc; t += 64) {
    float v = x[((long)b * SEQc + t) * CINc + c];
    a += v; a2 += v * v;
  }
  s1[threadIdx.x] = a; s2[threadIdx.x] = a2;
  __syncthreads();
  for (int s = 32; s > 0; s >>= 1) {
    if ((int)threadIdx.x < s) { s1[threadIdx.x] += s1[threadIdx.x + s];
                                s2[threadIdx.x] += s2[threadIdx.x + s]; }
    __syncthreads();
  }
  if (threadIdx.x == 0) {
    float m = s1[0] / SEQc;
    float var = s2[0] / SEQc - m * m;
    ms[bc] = m;
    ms[224 + bc] = sqrtf(var + 1e-5f);
  }
}

__global__ void embed_k(const float* x, const float* xmark, const float* tokw,
                        const float* tempw, const float* ms, float* emb) {
  long i = (long)blockIdx.x * blockDim.x + threadIdx.x;
  if (i >= (long)Bc * SEQc * DMc) return;
  int d = (int)(i % DMc);
  long bt = i / DMc;
  int t = (int)(bt % SEQc);
  int b = (int)(bt / SEQc);
  float acc = 0.f;
  for (int c = 0; c < CINc; ++c) {
    float m = ms[b * CINc + c], sd = ms[224 + b * CINc + c];
#pragma unroll
    for (int kk = 0; kk < 3; ++kk) {
      int tt = t + kk - 1;
      tt = (tt + SEQc) % SEQc;                       // circular pad
      float xe = (x[((long)b * SEQc + tt) * CINc + c] - m) / sd;
      acc += tokw[(d * CINc + c) * 3 + kk] * xe;
    }
  }
#pragma unroll
  for (int mi = 0; mi < 4; ++mi)
    acc += xmark[((long)b * SEQc + t) * 4 + mi] * tempw[d * 4 + mi];
  int half = d >> 1;
  float div = __expf(-(float)(2 * half) * (logf(10000.f) / (float)DMc));
  float arg = (float)t * div;
  acc += (d & 1) ? __cosf(arg) : __sinf(arg);
  emb[i] = acc;
}

// STFT: per (b, channel) reflect-pad series -> 85 frames * hann -> 64-pt DFT
__global__ void stft_k(const float* enc, bf16_t* Sr, bf16_t* Si) {
  __shared__ float xp[ALLc + NFFTc];   // 736
  __shared__ float win[64], ct[64], st[64];
  int b = blockIdx.x / DMc;
  int n = blockIdx.x % DMc;
  int tid = threadIdx.x;
  for (int i = tid; i < ALLc + NFFTc; i += blockDim.x) {
    int a;
    if (i < 32) a = 32 - i;             // reflect left
    else if (i < 704) a = i - 32;       // body
    else a = 1374 - i;                  // reflect right
    xp[i] = enc[((long)b * ALLc + a) * DMc + n];
  }
  if (tid < 64) {
    float ang = 2.f * 3.14159265358979323846f * (float)tid / 64.f;
    win[tid] = 0.5f - 0.5f * __cosf(ang);
    ct[tid] = __cosf(ang);
    st[tid] = __sinf(ang);
  }
  __syncthreads();
  long base = ((long)b * DMc + n) * (long)HWc;
  for (int o = tid; o < HWc; o += blockDim.x) {
    int q = o / NFc, f = o - (o / NFc) * NFc;
    float ar = 0.f, ai = 0.f;
    int s0 = f * 8;
#pragma unroll 4
    for (int j = 0; j < 64; ++j) {
      float xv = xp[s0 + j] * win[j];
      int idx = (q * j) & 63;
      ar += xv * ct[idx];
      ai -= xv * st[idx];
    }
    Sr[base + o] = (__bf16)(ar * 0.125f);   // ortho norm 1/sqrt(64)
    Si[base + o] = (__bf16)(ai * 0.125f);
  }
}

// mean over branches (/3), l0 linear, mean over freq (/64)
__global__ void freq_reduce_k(const float* sr, const float* si, float* y,
                              const float* l0w, const float* l0b) {
  long i = (long)blockIdx.x * blockDim.x + threadIdx.x;
  if (i >= (long)Bc * DMc * NFc) return;
  long bn = i / NFc;
  int w = (int)(i % NFc);
  long base = bn * (long)HWc + w;
  float ar = 0.f, ai = 0.f;
  for (int xq = 0; xq < NFFTc; ++xq) {
    ar += sr[base + (long)xq * NFc];
    ai += si[base + (long)xq * NFc];
  }
  y[i] = (l0w[0] * ar + l0w[1] * ai) * (1.f / (3.f * 64.f)) + l0b[0];
}

__global__ void lnorm_k(const float* xin, float* xout, const float* g,
                        const float* bta) {
  long base = (long)blockIdx.x * DMc;   // blocks = B*ALL
  int d = threadIdx.x;                  // 64
  __shared__ float s1[64], s2[64];
  float v = xin[base + d];
  s1[d] = v; s2[d] = v * v;
  __syncthreads();
  for (int s = 32; s > 0; s >>= 1) {
    if (d < s) { s1[d] += s1[d + s]; s2[d] += s2[d + s]; }
    __syncthreads();
  }
  float m = s1[0] / (float)DMc;
  float var = s2[0] / (float)DMc - m * m;
  xout[base + d] = (v - m) * rsqrtf(var + 1e-5f) * g[d] + bta[d];
}

// conv1d over d_model axis: 672ch -> 672ch, k=3, stride 2, pad 1
__global__ void cvconv_k(const float* enc, const float* w, const float* bvec,
                         float* h) {
  long i = (long)blockIdx.x * blockDim.x + threadIdx.x;
  if (i >= (long)Bc * ALLc * 32) return;
  int j = (int)(i % 32);
  long bo = i / 32;
  int o = (int)(bo % ALLc);
  int b = (int)(bo / ALLc);
  float acc = bvec[o];
  const float* wrow = w + (long)o * ALLc * 3;
  const float* eb = enc + (long)b * ALLc * DMc;
  int p = 2 * j - 1;
  for (int c = 0; c < ALLc; ++c) {
#pragma unroll
    for (int kk = 0; kk < 3; ++kk) {
      int pp = p + kk;
      if (pp >= 0 && pp < DMc) acc += wrow[c * 3 + kk] * eb[(long)c * DMc + pp];
    }
  }
  h[i] = acc;
}

__global__ void seqout_k(const float* h, const float* l1w, const float* l1b,
                         float* out2) {
  long i = (long)blockIdx.x * blockDim.x + threadIdx.x;
  if (i >= (long)Bc * ALLc) return;
  float acc = l1b[0];
  const float* hr = h + i * 32;
#pragma unroll 8
  for (int j = 0; j < 32; ++j) acc += hr[j] * l1w[j];
  out2[i] = acc;
}

__global__ void denorm_k(const float* dec, const float* ms, float* out1) {
  long i = (long)blockIdx.x * blockDim.x + threadIdx.x;
  if (i >= (long)Bc * SEQc * CINc) return;       // PRED == SEQ == 336
  int c = (int)(i % CINc);
  long bt = i / CINc;
  int t = (int)(bt % SEQc);
  int b = (int)(bt / SEQc);
  float v = dec[((long)b * ALLc + (SEQc + t)) * CINc + c];
  out1[i] = v * ms[224 + b * CINc + c] + ms[b * CINc + c];
}

// ---------------------------------------------------------------------------
// Launch
// ---------------------------------------------------------------------------
extern "C" void kernel_launch(void* const* d_in, const int* in_sizes, int n_in,
                              void* d_out, int out_size, void* d_ws,
                              size_t ws_size, hipStream_t stream) {
  const float* x_enc  = (const float*)d_in[0];
  const float* x_mark = (const float*)d_in[1];
  const float* tok_w  = (const float*)d_in[4];
  const float* temp_w = (const float*)d_in[5];
  const float* pl_w   = (const float*)d_in[6];
  const float* pl_b   = (const float*)d_in[7];
  const float* ln_g   = (const float*)d_in[8];
  const float* ln_b   = (const float*)d_in[9];
  const float* c0_wr  = (const float*)d_in[10];
  const float* c0_wi  = (const float*)d_in[11];
  const float* c0_br  = (const float*)d_in[12];
  const float* c0_bi  = (const float*)d_in[13];
  const float* cA_wr  = (const float*)d_in[14];
  const float* cA_wi  = (const float*)d_in[15];
  const float* cA_br  = (const float*)d_in[16];
  const float* cA_bi  = (const float*)d_in[17];
  const float* cB_wr  = (const float*)d_in[18];
  const float* cB_wi  = (const float*)d_in[19];
  const float* cB_br  = (const float*)d_in[20];
  const float* cB_bi  = (const float*)d_in[21];
  const float* l0_w   = (const float*)d_in[22];
  const float* l0_b   = (const float*)d_in[23];
  const float* l2_w   = (const float*)d_in[24];
  const float* l2_b   = (const float*)d_in[25];
  const float* cv_w   = (const float*)d_in[26];
  const float* cv_b   = (const float*)d_in[27];
  const float* l1_w   = (const float*)d_in[28];
  const float* l1_b   = (const float*)d_in[29];
  const float* proj_w = (const float*)d_in[30];
  const float* proj_b = (const float*)d_in[31];

  size_t off = 0;
  auto take = [&](size_t bytes) {
    void* p = (char*)d_ws + off;
    off += (bytes + 255) & ~(size_t)255;
    return p;
  };
  float* ms    = (float*)take(448 * sizeof(float));
  float* emb   = (float*)take((size_t)Bc * SEQc * DMc * sizeof(float));
  float* encA  = (float*)take((size_t)Bc * ALLc * DMc * sizeof(float));
  float* encB  = (float*)take((size_t)Bc * ALLc * DMc * sizeof(float));
  bf16_t* Sr   = (bf16_t*)take(SBIG * sizeof(bf16_t));  // also reused as hr
  bf16_t* Si   = (bf16_t*)take(SBIG * sizeof(bf16_t));  // also reused as hi
  bf16_t* Ar   = (bf16_t*)take(SBIG * sizeof(bf16_t));
  bf16_t* Ai   = (bf16_t*)take(SBIG * sizeof(bf16_t));
  bf16_t* Gr   = (bf16_t*)take((size_t)Bc * DMc * 64 * 64 * sizeof(bf16_t));
  bf16_t* Gi   = (bf16_t*)take((size_t)Bc * DMc * 64 * 64 * sizeof(bf16_t));
  float* SumR  = (float*)take(SBIG * sizeof(float));
  float* SumI  = (float*)take(SBIG * sizeof(float));
  float* yfreq = (float*)take((size_t)Bc * DMc * NFc * sizeof(float));
  float* dec   = (float*)take((size_t)Bc * ALLc * CINc * sizeof(float));
  float* hcv   = (float*)take((size_t)Bc * ALLc * 32 * sizeof(float));

  float* out1 = (float*)d_out;                    // [32, 336, 7]
  float* out2 = out1 + (size_t)Bc * SEQc * CINc;  // [32, 672]

  // ---- normalize stats + embedding + predict_linear -> encA [B,672,64] ----
  meanstd_k<<<Bc * CINc, 64, 0, stream>>>(x_enc, ms);
  {
    long tot = (long)Bc * SEQc * DMc;
    embed_k<<<(unsigned)((tot + 255) / 256), 256, 0, stream>>>(
        x_enc, x_mark, tok_w, temp_w, ms, emb);
  }
  gemm_dual<float, float><<<dim3(42 * 4, Bc), 32, 0, stream>>>(
      pl_w, emb, nullptr, nullptr, encA, nullptr, pl_b,
      ALLc, DMc, SEQc, SEQc, DMc, DMc,
      0L, (long)SEQc * DMc, (long)ALLc * DMc, 0.f, 0, 1, 0);

  const int BN = Bc * DMc;  // 2048 batched (b, channel) matrices

  for (int l = 0; l < 2; ++l) {
    // ---- STFT -> Sr/Si [B,DM,64,85] (bf16) ----
    stft_k<<<BN, 256, 0, stream>>>(encA, Sr, Si);

    // ---- gates: g_r = sigmoid(Sr Sr^T - Si Si^T); g_i = sigmoid(Sr Si^T + Si Sr^T)
    gemm_dual<bf16_t, bf16_t><<<dim3(16, BN), 32, 0, stream>>>(
        Sr, Sr, Si, Si, nullptr, Gr, nullptr, 64, 64, NFc, NFc, NFc, 64,
        (long)HWc, (long)HWc, 4096L, -1.f, 1, 0, 1);
    gemm_dual<bf16_t, bf16_t><<<dim3(16, BN), 32, 0, stream>>>(
        Sr, Si, Si, Sr, nullptr, Gi, nullptr, 64, 64, NFc, NFc, NFc, 64,
        (long)HWc, (long)HWc, 4096L, +1.f, 1, 0, 1);

    // ---- attention: ar = g_r Sr - g_i Si; ai = g_r Si + g_i Sr ----
    gemm_dual<bf16_t, bf16_t><<<dim3(4 * 6, BN), 32, 0, stream>>>(
        Gr, Sr, Gi, Si, nullptr, Ar, nullptr, 64, NFc, 64, 64, NFc, NFc,
        4096L, (long)HWc, (long)HWc, -1.f, 0, 0, 0);
    gemm_dual<bf16_t, bf16_t><<<dim3(4 * 6, BN), 32, 0, stream>>>(
        Gr, Si, Gi, Sr, nullptr, Ai, nullptr, 64, NFc, 64, 64, NFc, NFc,
        4096L, (long)HWc, (long)HWc, +1.f, 0, 0, 0);

    // ---- Inception branches accumulate into SumR/SumI ----
    hipMemsetAsync(SumR, 0, SBIG * sizeof(float), stream);
    hipMemsetAsync(SumI, 0, SBIG * sizeof(float), stream);
    dim3 cgrid(4 * (HWc / 32), Bc);  // (4*170, 32)

    // branch 0: 1x1 complex conv + relu -> sums
    cconv<<<cgrid, 32, 0, stream>>>(Ar, Ai, c0_wr + l * 4096, c0_wi + l * 4096,
                                    c0_br + l * 64, c0_bi + l * 64,
                                    nullptr, nullptr, SumR, SumI, 1, 1, 1);
    // branches 1,2: 3x3 dilated complex conv pairs (hr/hi reuse Sr/Si bufs)
    for (int k = 1; k <= 2; ++k) {
      long woff = (long)(l * 2 + (k - 1)) * DMc * DMc * 9;
      long boff = (long)(l * 2 + (k - 1)) * DMc;
      cconv<<<cgrid, 32, 0, stream>>>(Ar, Ai, cA_wr + woff, cA_wi + woff,
                                      cA_br + boff, cA_bi + boff,
                                      Sr, Si, nullptr, nullptr, 9, k, 0);
      cconv<<<cgrid, 32, 0, stream>>>(Sr, Si, cB_wr + woff, cB_wi + woff,
                                      cB_br + boff, cB_bi + boff,
                                      nullptr, nullptr, SumR, SumI, 9, k, 1);
    }

    // ---- branch mean + l0 + freq mean -> yfreq [B,DM,85] ----
    {
      long tot = (long)BN * NFc;
      freq_reduce_k<<<(unsigned)((tot + 255) / 256), 256, 0, stream>>>(
          SumR, SumI, yfreq, l0_w + l * 2, l0_b + l);
    }

    // ---- l2: [672 x 85] @ yfreq^T -> encB [B,672,64], then LayerNorm -> encA
    gemm_dual<float, float><<<dim3(42 * 4, Bc), 32, 0, stream>>>(
        l2_w + (long)l * ALLc * NFc, yfreq, nullptr, nullptr, encB, nullptr,
        l2_b + l * ALLc, ALLc, DMc, NFc, NFc, NFc, DMc,
        0L, (long)DMc * NFc, (long)ALLc * DMc, 0.f, 1, 1, 0);
    lnorm_k<<<Bc * ALLc, 64, 0, stream>>>(encB, encA, ln_g, ln_b);
  }

  // ---- seq_series head: conv1d(672->672,k3,s2) over d_model, then l1 ----
  {
    long tot = (long)Bc * ALLc * 32;
    cvconv_k<<<(unsigned)((tot + 255) / 256), 256, 0, stream>>>(encA, cv_w,
                                                               cv_b, hcv);
    long tot2 = (long)Bc * ALLc;
    seqout_k<<<(unsigned)((tot2 + 255) / 256), 256, 0, stream>>>(hcv, l1_w,
                                                                 l1_b, out2);
  }

  // ---- projection + de-normalization -> out1 ----
  gemm_dual<float, float><<<dim3(42 * 1, Bc), 32, 0, stream>>>(
      encA, proj_w, nullptr, nullptr, dec, nullptr, proj_b,
      ALLc, CINc, DMc, DMc, DMc, CINc,
      (long)ALLc * DMc, 0L, (long)ALLc * CINc, 0.f, 1, 2, 0);
  {
    long tot = (long)Bc * SEQc * CINc;
    denorm_k<<<(unsigned)((tot + 255) / 256), 256, 0, stream>>>(dec, ms, out1);
  }
}
